// SequenceAttention_71038759076085
// MI455X (gfx1250) — matrix operand order repaired and checked by reading, not compiled, not run on hardware
//
#include <hip/hip_runtime.h>

// ---------------------------------------------------------------------------
// MI455X (gfx1250) sequence-attention pipeline, bf16 WMMA + f32 accumulate.
//   K0: one-time f32->bf16 converts / transposes (so GEMM staging is pure copy)
//   K1: qp(bf16)   = query @ w_a                           (16384x512x512)
//   K2: simT(f16)  = (qp @ src^T)^T  + online row (m, l)   (8 x 2048x2048x512)
//   K3: out(f32)   = P^T @ src,  P = exp(sim - m) / l      (8 x 2048x512x2048)
// GEMMs: 8 waves/WG, WG tile 64x128, K-step 32, V_WMMA_F32_16X16X32_BF16,
// double-buffered LDS fed by GLOBAL_LOAD_ASYNC_TO_LDS_B128 (A tiles) and the
// Tensor Data Mover (B tiles, one tensor_load_to_lds per wave0 per K-step).
// ---------------------------------------------------------------------------

typedef __bf16    bf16;
typedef _Float16  f16;
typedef __attribute__((ext_vector_type(8)))  bf16  v8bf;
typedef __attribute__((ext_vector_type(16))) bf16  v16bf;
typedef __attribute__((ext_vector_type(8)))  f16   v8h;
typedef __attribute__((ext_vector_type(8)))  float v8f;
typedef __attribute__((ext_vector_type(4)))  float v4f;
typedef __attribute__((ext_vector_type(4)))  int   v4i;

#define B_  8
#define T_  2048
#define D_  512
#define BM  64      // WG tile M
#define BN  128     // WG tile N
#define BK  32      // K step (bf16 WMMA K)
#define LDT 40      // padded LDS row stride in halfs (80B) to spread banks

#define GAS __attribute__((address_space(1)))
#define LAS __attribute__((address_space(3)))

typedef GAS v4i gas_v4i;
typedef LAS v4i las_v4i;

// ------------------- gfx1250 async / TDM feature probes --------------------
#if defined(__has_builtin)
#  if __has_builtin(__builtin_amdgcn_global_load_async_to_lds_b128)
#    define HAS_ASYNC 1
#  endif
#  if __has_builtin(__builtin_amdgcn_tensor_load_to_lds)
#    define HAS_TDM 1
#  endif
#  if __has_builtin(__builtin_amdgcn_s_wait_asynccnt)
#    define HAS_WAITA 1
#  endif
#endif
#ifndef HAS_ASYNC
#define HAS_ASYNC 0
#endif
#ifndef HAS_TDM
#define HAS_TDM 0
#endif
#ifndef HAS_WAITA
#define HAS_WAITA 0
#endif

#if HAS_ASYNC
#  if HAS_WAITA
#    define WAIT_ASYNC(n) __builtin_amdgcn_s_wait_asynccnt((unsigned short)(n))
#  else
#    define WAIT_ASYNC(n) asm volatile("s_wait_asynccnt %0" ::"i"(n) : "memory")
#  endif
#  define A_PEND 1
#else
#  define WAIT_ASYNC(n) do {} while (0)
#  define A_PEND 0
#endif

#if HAS_TDM
// Unconditional: waves that issued no TDM op have TENSORcnt==0 and pass
// through; the issuing wave (wave 0) waits.  Avoids exec-mask churn.
#  define WAIT_TENSOR(n) __builtin_amdgcn_s_wait_tensorcnt((unsigned short)(n))
#  define B_PEND 0
#else
#  define WAIT_TENSOR(n) do {} while (0)
#  if HAS_ASYNC
#    define B_PEND 2
#  else
#    define B_PEND 0
#  endif
#endif

#define PEND_AB (A_PEND + B_PEND)
#define PEND_B  (B_PEND)

// 16-byte global -> LDS copy: async if available, else synchronous.
static __device__ __forceinline__ void async_cp16(const void* g, void* l) {
#if HAS_ASYNC
  __builtin_amdgcn_global_load_async_to_lds_b128((gas_v4i*)g, (las_v4i*)l, 0, 0);
#else
  *(v8bf*)l = *(const v8bf*)g;
#endif
}

// A tile: 64 rows x 32 halfs; one b128 per thread.
static __device__ __forceinline__ void issue_A_bf16(bf16* ls, const bf16* g,
                                                    int ldg, int tid) {
  int r = tid >> 2, c = (tid & 3) * 8;
  async_cp16(g + (size_t)r * ldg + c, ls + r * LDT + c);
}

#if HAS_TDM
typedef __attribute__((ext_vector_type(4))) unsigned int u32x4;
typedef __attribute__((ext_vector_type(8))) int          i32x8;
typedef __attribute__((ext_vector_type(4))) int          i32x4;

static __device__ __forceinline__ unsigned lds_off(const void* p) {
  return (unsigned)(size_t)(LAS const void*)p;
}

// One TDM descriptor: 128x32-element bf16 tile, row stride ldg elements,
// LDS rows padded 64B data + 16B pad (pad_interval=16 DW, pad_amount=4 DW).
static __device__ __forceinline__ void tdm_load_tile(const bf16* g, unsigned lds,
                                                     int ldg, int tdim1) {
  unsigned long long ga = (unsigned long long)(size_t)g;
  u32x4 g0;
  g0[0] = 1u;                                            // count=1, user D#
  g0[1] = lds;                                           // lds_addr
  g0[2] = (unsigned)ga;                                  // global_addr[31:0]
  g0[3] = (unsigned)((ga >> 32) & 0x01ffffffu) | (2u << 30);  // [56:32] + type=2
  unsigned dim0 = (unsigned)ldg;                         // tensor_dim0
  unsigned dim1 = (unsigned)tdim1;                       // tensor_dim1
  i32x8 g1;
  g1[0] = (int)((1u << 16)        // data_size = 2 bytes
              | (1u << 20)        // pad_enable
              | (3u << 22)        // pad_interval: 16 DWORDs
              | (3u << 25));      // pad_amount:   4 DWORDs
  g1[1] = (int)((dim0 & 0xffffu) << 16);
  g1[2] = (int)((dim0 >> 16) | ((dim1 & 0xffffu) << 16));
  g1[3] = (int)((dim1 >> 16) | (32u << 16));             // tile_dim0 = 32
  g1[4] = (int)128u;                                     // tile_dim1 = 128
  g1[5] = (int)dim0;                                     // dim0_stride[31:0]
  g1[6] = 0;
  g1[7] = 0;
  i32x4 z4 = {0, 0, 0, 0};
  i32x8 z8 = {0, 0, 0, 0, 0, 0, 0, 0};
  __builtin_amdgcn_tensor_load_to_lds(g0, g1, z4, z4, z8, 0);
}
#endif

// B tile: 128 rows x 32 halfs -> ls[n][k].
static __device__ __forceinline__ void stage_B(bf16* ls, const bf16* g, int ldg,
                                               int tdim1, int tid, int wv) {
#if HAS_TDM
  (void)tid;
  if (wv == 0) tdm_load_tile(g, lds_off(ls), ldg, tdim1);
#else
  (void)wv; (void)tdim1;
  int r = tid >> 1, c = (tid & 1) * 16;
  async_cp16(g + (size_t)r * ldg + c, ls + r * LDT + c);
  async_cp16(g + (size_t)r * ldg + c + 8, ls + r * LDT + c + 8);
#endif
}

// --------------------------- WMMA core -------------------------------------
static __device__ __forceinline__ v8f wmma_bf(v16bf a, v16bf b, v8f c) {
  return __builtin_amdgcn_wmma_f32_16x16x32_bf16(false, a, false, b,
                                                 (short)0, c, false, false);
}

static __device__ __forceinline__ v16bf frag2(const bf16* p0, const bf16* p1) {
  union { v8bf h[2]; v16bf v; } u;
  u.h[0] = *(const v8bf*)p0;
  u.h[1] = *(const v8bf*)p1;
  return u.v;
}

static __device__ __forceinline__ void wave_mma(const bf16* lsA, const bf16* lsB,
                                                int mw, int nw, int lr, int hi,
                                                v8f cacc[4]) {
  const bf16* ap = &lsA[(mw * 16 + lr) * LDT + hi * 8];
  v16bf a = frag2(ap, ap + 16);
#pragma unroll
  for (int t = 0; t < 4; ++t) {
    const bf16* bp = &lsB[(nw * 64 + t * 16 + lr) * LDT + hi * 8];
    v16bf b = frag2(bp, bp + 16);
    cacc[t] = wmma_bf(a, b, cacc[t]);
  }
}

// A tile for K3: simT f16 [j][i] with transform exp(s - m_i) * invl_i.
static __device__ __forceinline__ void stage_A_p(bf16* ls, const f16* g, int ldg,
                                                 const float* mv, const float* ilv,
                                                 int tid) {
  int r = tid >> 2, c = (tid & 3) * 8;
  v8h s  = *(const v8h*)(g + (size_t)r * ldg + c);
  v4f m0 = *(const v4f*)(mv + c);
  v4f m1 = *(const v4f*)(mv + c + 4);
  v4f i0 = *(const v4f*)(ilv + c);
  v4f i1 = *(const v4f*)(ilv + c + 4);
  v8bf h;
  h[0] = (bf16)(__expf((float)s[0] - m0[0]) * i0[0]);
  h[1] = (bf16)(__expf((float)s[1] - m0[1]) * i0[1]);
  h[2] = (bf16)(__expf((float)s[2] - m0[2]) * i0[2]);
  h[3] = (bf16)(__expf((float)s[3] - m0[3]) * i0[3]);
  h[4] = (bf16)(__expf((float)s[4] - m1[0]) * i1[0]);
  h[5] = (bf16)(__expf((float)s[5] - m1[1]) * i1[1]);
  h[6] = (bf16)(__expf((float)s[6] - m1[2]) * i1[2]);
  h[7] = (bf16)(__expf((float)s[7] - m1[3]) * i1[3]);
  *(v8bf*)&ls[r * LDT + c] = h;
}

// --------------------- K0a: flat f32 -> bf16 convert ------------------------
__global__ __launch_bounds__(256)
void k0_conv(const float* __restrict__ in, bf16* __restrict__ outp) {
  size_t t = (size_t)blockIdx.x * 256 + threadIdx.x;
  const float* p = in + t * 8;
  v4f x0 = *(const v4f*)p;
  v4f x1 = *(const v4f*)(p + 4);
  v8bf h;
  h[0]=(bf16)x0[0]; h[1]=(bf16)x0[1]; h[2]=(bf16)x0[2]; h[3]=(bf16)x0[3];
  h[4]=(bf16)x1[0]; h[5]=(bf16)x1[1]; h[6]=(bf16)x1[2]; h[7]=(bf16)x1[3];
  *(v8bf*)(outp + t * 8) = h;
}

// ------------- K0b: transpose+convert: in[R][C] f32 -> out[C][R] bf16 -------
__global__ __launch_bounds__(256)
void k0_tconv(const float* __restrict__ in, bf16* __restrict__ outp,
              int R, int C) {
  __shared__ float ls[32][33];
  int b = blockIdx.z;
  const float* ip = in + (size_t)b * R * C;
  bf16* op = outp + (size_t)b * R * C;
  int r0 = blockIdx.x * 32, c0 = blockIdx.y * 32;
  int tid = threadIdx.x;
  int tr = tid >> 3, tc = (tid & 7) * 4;
  v4f x = *(const v4f*)(ip + (size_t)(r0 + tr) * C + c0 + tc);
  ls[tr][tc + 0] = x[0]; ls[tr][tc + 1] = x[1];
  ls[tr][tc + 2] = x[2]; ls[tr][tc + 3] = x[3];
  __syncthreads();
  union { bf16 h[4]; uint2 u; } pk;
  pk.h[0] = (bf16)ls[tc + 0][tr];
  pk.h[1] = (bf16)ls[tc + 1][tr];
  pk.h[2] = (bf16)ls[tc + 2][tr];
  pk.h[3] = (bf16)ls[tc + 3][tr];
  *(uint2*)(op + (size_t)(c0 + tr) * R + r0 + tc) = pk.u;
}

// --------------------------- K1: qp = query @ w_a --------------------------
__global__ __launch_bounds__(256)
void k1_proj(const bf16* __restrict__ qbf, const bf16* __restrict__ waT,
             bf16* __restrict__ qp) {
  __shared__ __align__(16) bf16 lsA[2][BM * LDT];
  __shared__ __align__(16) bf16 lsB[2][BN * LDT];
  const int tid = threadIdx.x, lane = tid & 31, wv = tid >> 5;
  const int mw = wv & 3, nw = wv >> 2;
  const int lr = lane & 15, hi = lane >> 4;
  const int mbase = blockIdx.x * BM;
  const int nbase = blockIdx.y * BN;
  const bf16* Ag = qbf + (size_t)mbase * D_;
  const bf16* Bg = waT + (size_t)nbase * D_;

  v8f cacc[4] = {};
  issue_A_bf16(lsA[0], Ag, D_, tid);
  stage_B(lsB[0], Bg, D_, D_, tid, wv);
  int buf = 0;
  const int NK = D_ / BK;
  for (int ks = 0; ks < NK; ++ks) {
    if (ks + 1 < NK) {
      int kn = (ks + 1) * BK;
      issue_A_bf16(lsA[buf ^ 1], Ag + kn, D_, tid);
      stage_B(lsB[buf ^ 1], Bg + kn, D_, D_, tid, wv);
      WAIT_ASYNC(PEND_AB);
      WAIT_TENSOR(1);
    } else {
      WAIT_ASYNC(0);
      WAIT_TENSOR(0);
    }
    __syncthreads();
    wave_mma(lsA[buf], lsB[buf], mw, nw, lr, hi, cacc);
    __syncthreads();
    buf ^= 1;
  }
#pragma unroll
  for (int t = 0; t < 4; ++t) {
    int col = nbase + nw * 64 + t * 16 + lr;
#pragma unroll
    for (int r = 0; r < 8; ++r) {
      int row = mbase + mw * 16 + hi * 8 + r;
      qp[(size_t)row * D_ + col] = (bf16)cacc[t][r];
    }
  }
}

// ----- K2: simT[b][j][i] = <qp[b,i], src[b,j]> (f16) + row stats (m, 1/l) ---
__global__ __launch_bounds__(256)
void k2_sim(const bf16* __restrict__ qp, const bf16* __restrict__ sbf,
            f16* __restrict__ simT, float* __restrict__ stat_m,
            float* __restrict__ stat_il) {
  __shared__ __align__(16) bf16 lsA[2][BM * LDT];
  __shared__ __align__(16) bf16 lsB[2][BN * LDT];
  __shared__ float redM[4 * 2 * 16];
  __shared__ float redL[4 * 2 * 16];
  const int tid = threadIdx.x, lane = tid & 31, wv = tid >> 5;
  const int mw = wv & 3, nw = wv >> 2;
  const int lr = lane & 15, hi = lane >> 4;
  const int b = blockIdx.y;
  const int ibase = blockIdx.x * BM;

  const bf16* Ag = qp  + (size_t)b * T_ * D_ + (size_t)ibase * D_;
  const bf16* Sg = sbf + (size_t)b * T_ * D_;
  f16*        PT = simT + (size_t)b * T_ * T_;

  const float NEGINF = -__builtin_inff();
  float m8[8], l8[8];
#pragma unroll
  for (int r = 0; r < 8; ++r) { m8[r] = NEGINF; l8[r] = 0.0f; }

  int buf = 0;
  const int NK = D_ / BK;
  for (int j0 = 0; j0 < T_; j0 += BN) {
    const bf16* Bg = Sg + (size_t)j0 * D_;
    v8f cacc[4] = {};
    issue_A_bf16(lsA[buf], Ag, D_, tid);
    stage_B(lsB[buf], Bg, D_, T_, tid, wv);
    for (int ks = 0; ks < NK; ++ks) {
      if (ks + 1 < NK) {
        int kn = (ks + 1) * BK;
        issue_A_bf16(lsA[buf ^ 1], Ag + kn, D_, tid);
        stage_B(lsB[buf ^ 1], Bg + kn, D_, T_, tid, wv);
        WAIT_ASYNC(PEND_AB);
        WAIT_TENSOR(1);
      } else {
        WAIT_ASYNC(0);
        WAIT_TENSOR(0);
      }
      __syncthreads();
      wave_mma(lsA[buf], lsB[buf], mw, nw, lr, hi, cacc);
      __syncthreads();
      buf ^= 1;
    }
    // store simT (each lane's 8 accumulator rows are contiguous i -> one b128)
#pragma unroll
    for (int t = 0; t < 4; ++t) {
      int j = j0 + nw * 64 + t * 16 + lr;
      union { f16 h[8]; uint4 u; } pk;
#pragma unroll
      for (int r = 0; r < 8; ++r) pk.h[r] = (f16)cacc[t][r];
      int i_lo = ibase + mw * 16 + hi * 8;
      *(uint4*)(PT + (size_t)j * T_ + i_lo) = pk.u;
    }
    // online (max, sumexp) per row
#pragma unroll
    for (int r = 0; r < 8; ++r) {
      float s0 = cacc[0][r], s1 = cacc[1][r], s2 = cacc[2][r], s3 = cacc[3][r];
      float tm = fmaxf(fmaxf(s0, s1), fmaxf(s2, s3));
      float mn = fmaxf(m8[r], tm);
      l8[r] = l8[r] * __expf(m8[r] - mn) + __expf(s0 - mn) + __expf(s1 - mn) +
              __expf(s2 - mn) + __expf(s3 - mn);
      m8[r] = mn;
    }
  }
  // reduce across the 16 lanes of each half (same rows), then across nw waves
#pragma unroll
  for (int r = 0; r < 8; ++r) {
    float m = m8[r], l = l8[r];
#pragma unroll
    for (int mask = 1; mask < 16; mask <<= 1) {
      float om = __shfl_xor(m, mask, 32);
      float ol = __shfl_xor(l, mask, 32);
      float mn = fmaxf(m, om);
      l = l * __expf(m - mn) + ol * __expf(om - mn);
      m = mn;
    }
    if (lr == 0) {
      int row = hi * 8 + r;
      redM[(mw * 2 + nw) * 16 + row] = m;
      redL[(mw * 2 + nw) * 16 + row] = l;
    }
  }
  __syncthreads();
  if (tid < 64) {
    int mw2 = tid >> 4, row = tid & 15;
    float ma = redM[(mw2 * 2 + 0) * 16 + row];
    float mb = redM[(mw2 * 2 + 1) * 16 + row];
    float mn = fmaxf(ma, mb);
    float l  = redL[(mw2 * 2 + 0) * 16 + row] * __expf(ma - mn) +
               redL[(mw2 * 2 + 1) * 16 + row] * __expf(mb - mn);
    int gi = ibase + mw2 * 16 + row;
    stat_m[(size_t)b * T_ + gi]  = mn;
    stat_il[(size_t)b * T_ + gi] = 1.0f / l;
  }
}

// ------------- K3: out[b][j][d] = sum_i P[i][j] * src[b][i][d] -------------
__global__ __launch_bounds__(256)
void k3_out(const f16* __restrict__ simT, const bf16* __restrict__ sTbf,
            const float* __restrict__ stat_m, const float* __restrict__ stat_il,
            float* __restrict__ out) {
  __shared__ __align__(16) bf16 lsA[2][BM * LDT];
  __shared__ __align__(16) bf16 lsB[2][BN * LDT];
  const int tid = threadIdx.x, lane = tid & 31, wv = tid >> 5;
  const int mw = wv & 3, nw = wv >> 2;
  const int lr = lane & 15, hi = lane >> 4;
  const int b = blockIdx.z;
  const int jbase = blockIdx.x * BM;
  const int dbase = blockIdx.y * BN;

  const f16*   Ag = simT + (size_t)b * T_ * T_ + (size_t)jbase * T_;
  const bf16*  Bg = sTbf + (size_t)b * D_ * T_ + (size_t)dbase * T_;
  const float* M  = stat_m  + (size_t)b * T_;
  const float* IL = stat_il + (size_t)b * T_;

  v8f cacc[4] = {};
  stage_A_p(lsA[0], Ag, T_, M, IL, tid);
  stage_B(lsB[0], Bg, T_, D_, tid, wv);
  int buf = 0;
  const int NK = T_ / BK;
  for (int ks = 0; ks < NK; ++ks) {
    if (ks + 1 < NK) {
      int kn = (ks + 1) * BK;
      stage_A_p(lsA[buf ^ 1], Ag + kn, T_, M + kn, IL + kn, tid);
      stage_B(lsB[buf ^ 1], Bg + kn, T_, D_, tid, wv);
      WAIT_ASYNC(PEND_B);
      WAIT_TENSOR(1);
    } else {
      WAIT_ASYNC(0);
      WAIT_TENSOR(0);
    }
    __syncthreads();
    wave_mma(lsA[buf], lsB[buf], mw, nw, lr, hi, cacc);
    __syncthreads();
    buf ^= 1;
  }
#pragma unroll
  for (int t = 0; t < 4; ++t) {
    int d = dbase + nw * 64 + t * 16 + lr;
#pragma unroll
    for (int r = 0; r < 8; ++r) {
      int j = jbase + mw * 16 + hi * 8 + r;
      out[((size_t)b * T_ + j) * D_ + d] = cacc[t][r];
    }
  }
}

// ------------------------------- launcher ----------------------------------
extern "C" void kernel_launch(void* const* d_in, const int* in_sizes, int n_in,
                              void* d_out, int out_size, void* d_ws, size_t ws_size,
                              hipStream_t stream) {
  (void)in_sizes; (void)n_in; (void)out_size; (void)ws_size;
  const float* source = (const float*)d_in[0];
  const float* query  = (const float*)d_in[1];
  const float* w_a    = (const float*)d_in[2];
  float* out = (float*)d_out;

  char* w = (char*)d_ws;
  bf16* qbf  = (bf16*)w; w += (size_t)B_ * T_ * D_ * sizeof(bf16);
  bf16* sbf  = (bf16*)w; w += (size_t)B_ * T_ * D_ * sizeof(bf16);
  bf16* sTbf = (bf16*)w; w += (size_t)B_ * T_ * D_ * sizeof(bf16);
  bf16* waT  = (bf16*)w; w += (size_t)D_ * D_ * sizeof(bf16);
  bf16* qp   = (bf16*)w; w += (size_t)B_ * T_ * D_ * sizeof(bf16);
  f16*  simT = (f16*)w;  w += (size_t)B_ * T_ * T_ * sizeof(f16);
  float* sm  = (float*)w; w += (size_t)B_ * T_ * sizeof(float);
  float* sil = (float*)w;

  dim3 blk(256);
  int nconv = (B_ * T_ * D_) / (8 * 256);
  k0_conv<<<nconv, blk, 0, stream>>>(query, qbf);
  k0_conv<<<nconv, blk, 0, stream>>>(source, sbf);
  k0_tconv<<<dim3(T_ / 32, D_ / 32, B_), blk, 0, stream>>>(source, sTbf, T_, D_);
  k0_tconv<<<dim3(D_ / 32, D_ / 32, 1), blk, 0, stream>>>(w_a, waT, D_, D_);
  k1_proj<<<dim3((B_ * T_) / BM, D_ / BN), blk, 0, stream>>>(qbf, waT, qp);
  k2_sim<<<dim3(T_ / BM, B_), blk, 0, stream>>>(qp, sbf, simT, sm, sil);
  k3_out<<<dim3(T_ / BM, D_ / BN, B_), blk, 0, stream>>>(simT, sTbf, sm, sil, out);
}
